// LoRWeBLoRAModule_87720412053638
// MI455X (gfx1250) — compile-verified, matrix operand order checked by reading, not run on hardware
//
#include <hip/hip_runtime.h>
#include <hip/hip_bf16.h>

// ---------------------------------------------------------------------------
// Problem constants (from the reference)
// ---------------------------------------------------------------------------
constexpr int B_   = 4;
constexpr int S_   = 4096;
constexpr int IN_  = 2048;
constexpr int OUT_ = 2048;
constexpr int NB_  = 32;    // N_BASIS
constexpr int KD_  = 128;   // KEYS_DIM
constexpr int LR_  = 4;     // LORA_DIM
constexpr float SCALE_ = 0.25f;  // MULTIPLIER * ALPHA / LORA_DIM

typedef __attribute__((ext_vector_type(16))) __bf16 v16bf;
typedef __attribute__((ext_vector_type(8)))  float  v8f;

// ---------------------------------------------------------------------------
// CDNA5 async global->LDS copy (ASYNCcnt-tracked). Probe round 2 showed the
// builtin exists with param0 = 'int __vector(4) addrspace(1)*' (non-const).
// ---------------------------------------------------------------------------
#if defined(__HIP_DEVICE_COMPILE__) && __has_builtin(__builtin_amdgcn_global_load_async_to_lds_b128)
#define USE_ASYNC_LDS 1
#else
#define USE_ASYNC_LDS 0
#endif

#if USE_ASYNC_LDS
typedef int v4i_vec __attribute__((vector_size(4 * sizeof(int))));
typedef __attribute__((address_space(1))) v4i_vec* g_v4i_ptr;
typedef __attribute__((address_space(3))) v4i_vec* l_v4i_ptr;

__device__ __forceinline__ void async_cp16(const void* g, void* l)
{
    __builtin_amdgcn_global_load_async_to_lds_b128(
        (g_v4i_ptr)g, (l_v4i_ptr)l, /*imm offset*/ 0, /*cpol*/ 0);
}

template <int N>
__device__ __forceinline__ void wait_asynccnt()
{
#if __has_builtin(__builtin_amdgcn_s_wait_asynccnt)
    __builtin_amdgcn_s_wait_asynccnt(N);
#else
    asm volatile("s_wait_asynccnt %0" :: "i"(N) : "memory");
#endif
}
#endif

// ---------------------------------------------------------------------------
// Kernel 1: pooled mean -> q -> logits -> softmax -> mixing[B, NB]
// One block per batch, 256 threads.
// ---------------------------------------------------------------------------
__global__ void mixing_kernel(const float* __restrict__ x,
                              const float* __restrict__ w_q,
                              const float* __restrict__ keys,
                              float* __restrict__ mixing)
{
    __shared__ float pooled[IN_];
    __shared__ float q[KD_];
    __shared__ float logits[NB_];

    const int b = blockIdx.x;
    const int t = threadIdx.x;  // 256 threads
    const float* xb = x + (size_t)b * S_ * IN_;

    float acc[IN_ / 256];
#pragma unroll
    for (int j = 0; j < IN_ / 256; ++j) acc[j] = 0.f;
    for (int s = 0; s < S_; ++s) {
        const float* row = xb + (size_t)s * IN_;
#pragma unroll
        for (int j = 0; j < IN_ / 256; ++j) acc[j] += row[t + 256 * j];
    }
#pragma unroll
    for (int j = 0; j < IN_ / 256; ++j) pooled[t + 256 * j] = acc[j] * (1.0f / S_);
    __syncthreads();

    if (t < KD_) {
        float s = 0.f;
        const float* wr = w_q + (size_t)t * IN_;
        for (int i = 0; i < IN_; ++i) s += wr[i] * pooled[i];
        q[t] = s;
    }
    __syncthreads();

    if (t < NB_) {
        float s = 0.f;
        const float* kr = keys + (size_t)t * KD_;
        for (int i = 0; i < KD_; ++i) s += kr[i] * q[i];
        logits[t] = s * 0.08838834764831845f;  // 128^-0.5
    }
    __syncthreads();

    if (t == 0) {
        float mx = -1e30f;
        for (int n = 0; n < NB_; ++n) mx = fmaxf(mx, logits[n]);
        float e[NB_];
        float sum = 0.f;
        for (int n = 0; n < NB_; ++n) { e[n] = __expf(logits[n] - mx); sum += e[n]; }
        float inv = 1.0f / sum;
        for (int n = 0; n < NB_; ++n) mixing[b * NB_ + n] = e[n] * inv;
    }
}

// ---------------------------------------------------------------------------
// Kernel 2: x fp32 -> bf16 (one-time; halves A-tile traffic in the GEMM)
// ---------------------------------------------------------------------------
__global__ void xcast_kernel(const float* __restrict__ x,
                             __hip_bfloat16* __restrict__ xb, size_t n)
{
    size_t i = (size_t)blockIdx.x * blockDim.x + threadIdx.x;
    size_t stride = (size_t)gridDim.x * blockDim.x;
    for (; i < n; i += stride) xb[i] = __float2bfloat16(x[i]);
}

// ---------------------------------------------------------------------------
// Kernel 3: W_eff_b[o,i] = w_org[o,i] + SCALE * sum_{n,r} m_bn up[n,r,o] down[n,i,r]
// ---------------------------------------------------------------------------
__global__ void weff_kernel(const float* __restrict__ w_org,
                            const float* __restrict__ down,    // [NB, IN, LR]
                            const float* __restrict__ up,      // [NB, LR, OUT]
                            const float* __restrict__ mixing,  // [B, NB]
                            __hip_bfloat16* __restrict__ weff) // [B, OUT, IN]
{
    __shared__ float Ds[NB_ * LR_][16];  // [nr][i_local]
    __shared__ float Us[NB_ * LR_][16];  // [nr][o_local], pre-scaled by mixing

    const int b  = blockIdx.z;
    const int i0 = blockIdx.x * 16;
    const int o0 = blockIdx.y * 16;
    const int tx = threadIdx.x, ty = threadIdx.y;
    const int t  = ty * 16 + tx;

    for (int e = t; e < NB_ * LR_ * 16; e += 256) {
        int nr = e >> 4, il = e & 15;
        int n = nr >> 2, r = nr & 3;
        Ds[nr][il] = down[((size_t)n * IN_ + (i0 + il)) * LR_ + r];
        Us[nr][il] = up[((size_t)n * LR_ + r) * OUT_ + (o0 + il)] * mixing[b * NB_ + n];
    }
    __syncthreads();

    float acc = 0.f;
#pragma unroll 8
    for (int k = 0; k < NB_ * LR_; ++k) acc += Ds[k][tx] * Us[k][ty];

    size_t idx = ((size_t)b * OUT_ + (o0 + ty)) * IN_ + (i0 + tx);
    weff[idx] = __float2bfloat16(w_org[(size_t)(o0 + ty) * IN_ + (i0 + tx)] + SCALE_ * acc);
}

// ---------------------------------------------------------------------------
// Kernel 4: out[b] = x_bf16[b] @ W_eff_b^T + bias      (bf16 WMMA, fp32 acc)
// 256 threads = 8 waves, 128x128 tile; wave owns 32x64 = 2x4 WMMA accums.
// Double-buffered LDS fed by GLOBAL_LOAD_ASYNC_TO_LDS_B128 when available.
// ---------------------------------------------------------------------------
#define BM 128
#define BN 128
#define KC 32

__global__ void __launch_bounds__(256)
gemm_kernel(const __hip_bfloat16* __restrict__ xb,    // [B, S, IN]
            const __hip_bfloat16* __restrict__ weff,  // [B, OUT, IN]
            const float* __restrict__ bias,           // [OUT]
            float* __restrict__ out)                  // [B, S, OUT]
{
    __shared__ __hip_bfloat16 As[2][BM][KC];  // x tiles:    [buf][s_local][k]
    __shared__ __hip_bfloat16 Bs[2][BN][KC];  // weff tiles: [buf][o_local][k]

    const int b  = blockIdx.z;
    const int o0 = blockIdx.x * BN;
    const int s0 = blockIdx.y * BM;
    const int t    = threadIdx.x;
    const int lane = t & 31;
    const int wave = t >> 5;      // 0..7
    const int wm   = wave & 3;    // row group: rows 32*wm .. +32
    const int wn   = wave >> 2;   // col group: cols 64*wn .. +64
    const int half = lane >> 4;
    const int l16  = lane & 15;

    const __hip_bfloat16* Ab = xb   + (size_t)b * S_   * IN_;
    const __hip_bfloat16* Bb = weff + (size_t)b * OUT_ * IN_;

    v8f acc[2][4];
#pragma unroll
    for (int i = 0; i < 2; ++i)
#pragma unroll
        for (int j = 0; j < 4; ++j) acc[i][j] = (v8f){0, 0, 0, 0, 0, 0, 0, 0};

    // Staging assignment: 2 threads per row, each covers 32 B (two b128s).
    const int arow  = t >> 1;
    const int acolb = (t & 1) * 32;  // byte offset within the 64 B row

    const char* gA = (const char*)(Ab + (size_t)(s0 + arow) * IN_) + acolb;
    const char* gB = (const char*)(Bb + (size_t)(o0 + arow) * IN_) + acolb;

    // Per-buffer compute on staged tiles (ISA wave32 16-bit A/B layouts:
    // lane -> row/col l16, VGPR v -> packed k-pair at
    // kb = (v&4?16:0) + half*8 + (v&3)*2; C: M = v + 8*half, N = l16).
    auto compute = [&](int buf) {
        union FragB { unsigned int u[8]; v16bf v; } fb[4];
#pragma unroll
        for (int tn = 0; tn < 4; ++tn) {
            int n = wn * 64 + tn * 16 + l16;
#pragma unroll
            for (int v = 0; v < 8; ++v) {
                int kb = ((v & 4) ? 16 : 0) + half * 8 + (v & 3) * 2;
                fb[tn].u[v] = *(const unsigned int*)&Bs[buf][n][kb];
            }
        }
#pragma unroll
        for (int tm = 0; tm < 2; ++tm) {
            union FragA { unsigned int u[8]; v16bf v; } fa;
            int m = wm * 32 + tm * 16 + l16;
#pragma unroll
            for (int v = 0; v < 8; ++v) {
                int kb = ((v & 4) ? 16 : 0) + half * 8 + (v & 3) * 2;
                fa.u[v] = *(const unsigned int*)&As[buf][m][kb];
            }
#pragma unroll
            for (int tn = 0; tn < 4; ++tn) {
                acc[tm][tn] = __builtin_amdgcn_wmma_f32_16x16x32_bf16(
                    false, fa.v, false, fb[tn].v, (short)0, acc[tm][tn],
                    false, false);
            }
        }
    };

    constexpr int NSTEP = IN_ / KC;  // 64

#if USE_ASYNC_LDS
    // Issue the 4 async b128s (A-half x2, B-half x2) for one K-chunk.
    auto issue = [&](int buf, int kbyte) {
        char* la = (char*)&As[buf][arow][0] + acolb;
        char* lb = (char*)&Bs[buf][arow][0] + acolb;
        async_cp16(gA + kbyte,      la);
        async_cp16(gA + kbyte + 16, la + 16);
        async_cp16(gB + kbyte,      lb);
        async_cp16(gB + kbyte + 16, lb + 16);
    };

    issue(0, 0);
    for (int s = 0; s < NSTEP; ++s) {
        const int cur = s & 1;
        if (s + 1 < NSTEP) {
            issue(cur ^ 1, (s + 1) * KC * 2);
            wait_asynccnt<4>();   // older (current-buffer) loads retired
        } else {
            wait_asynccnt<0>();
        }
        __syncthreads();          // publish current buffer to all waves
        compute(cur);
        __syncthreads();          // everyone done reading before overwrite
    }
#else
    // Fallback: sync staging via VGPRs (proven path).
    for (int s = 0; s < NSTEP; ++s) {
        const int cur = s & 1;
        const int kbyte = s * KC * 2;
        uint4 a0 = *(const uint4*)(gA + kbyte);
        uint4 a1 = *(const uint4*)(gA + kbyte + 16);
        uint4 b0 = *(const uint4*)(gB + kbyte);
        uint4 b1 = *(const uint4*)(gB + kbyte + 16);
        char* la = (char*)&As[cur][arow][0] + acolb;
        char* lb = (char*)&Bs[cur][arow][0] + acolb;
        *(uint4*)(la)      = a0;
        *(uint4*)(la + 16) = a1;
        *(uint4*)(lb)      = b0;
        *(uint4*)(lb + 16) = b1;
        __syncthreads();
        compute(cur);
        __syncthreads();
    }
#endif

    // Epilogue: add bias, write fp32.
    float* outb = out + (size_t)b * S_ * OUT_;
#pragma unroll
    for (int tm = 0; tm < 2; ++tm) {
#pragma unroll
        for (int tn = 0; tn < 4; ++tn) {
            int col = o0 + wn * 64 + tn * 16 + l16;
            float bv = bias[col];
#pragma unroll
            for (int v = 0; v < 8; ++v) {
                int row = s0 + wm * 32 + tm * 16 + v + half * 8;
                outb[(size_t)row * OUT_ + col] = acc[tm][tn][v] + bv;
            }
        }
    }
}

// ---------------------------------------------------------------------------
// Launch
// ---------------------------------------------------------------------------
extern "C" void kernel_launch(void* const* d_in, const int* in_sizes, int n_in,
                              void* d_out, int out_size, void* d_ws, size_t ws_size,
                              hipStream_t stream)
{
    const float* x     = (const float*)d_in[0];  // [B,S,IN]
    const float* w_org = (const float*)d_in[1];  // [OUT,IN]
    const float* b_org = (const float*)d_in[2];  // [OUT]
    const float* w_q   = (const float*)d_in[3];  // [KD,IN]
    const float* keys  = (const float*)d_in[4];  // [NB,KD]
    const float* down  = (const float*)d_in[5];  // [NB,IN,LR]
    const float* up    = (const float*)d_in[6];  // [NB,LR,OUT]
    float* out = (float*)d_out;                  // [B,S,OUT]

    char* ws = (char*)d_ws;
    float* mixing        = (float*)ws;                                   // B*NB f32
    __hip_bfloat16* xbf  = (__hip_bfloat16*)(ws + 4096);                 // B*S*IN bf16 (64 MB)
    __hip_bfloat16* weff = (__hip_bfloat16*)(ws + 4096 +
                               (size_t)B_ * S_ * IN_ * sizeof(__hip_bfloat16)); // B*OUT*IN bf16 (32 MB)

    mixing_kernel<<<B_, 256, 0, stream>>>(x, w_q, keys, mixing);
    xcast_kernel<<<2048, 256, 0, stream>>>(x, xbf, (size_t)B_ * S_ * IN_);
    weff_kernel<<<dim3(IN_ / 16, OUT_ / 16, B_), dim3(16, 16), 0, stream>>>(
        w_org, down, up, mixing, weff);
    gemm_kernel<<<dim3(OUT_ / BN, S_ / BM, B_), 256, 0, stream>>>(xbf, weff, b_org, out);
}
